// MolGNN_76390288326656
// MI455X (gfx1250) — matrix operand order repaired and checked by reading, not compiled, not run on hardware
//
#include <hip/hip_runtime.h>

typedef __bf16 bf16;
typedef __attribute__((ext_vector_type(16))) bf16 v16bf;
typedef __attribute__((ext_vector_type(8)))  float v8f;
typedef __attribute__((address_space(3)))    bf16 lds_bf16_t;

#define HDIM 256

__device__ __forceinline__ void atomAddF(float* p, float v) {
  unsafeAtomicAdd(p, v);
}

// ---------------------------------------------------------------- utilities
__global__ void cvt_f32_bf16(const float* __restrict__ src, bf16* __restrict__ dst, int n) {
  int i = blockIdx.x * blockDim.x + threadIdx.x;
  int stride = gridDim.x * blockDim.x;
  for (; i < n; i += stride) dst[i] = (bf16)src[i];
}

__global__ void zero_f32(float* __restrict__ p, int n4) {
  int i = blockIdx.x * blockDim.x + threadIdx.x;
  int stride = gridDim.x * blockDim.x;
  float4 z = make_float4(0.f, 0.f, 0.f, 0.f);
  float4* p4 = (float4*)p;
  for (; i < n4; i += stride) p4[i] = z;
}

// --------------------------------------------- node embedding sum + LayerNorm
__global__ __launch_bounds__(256)
void node_embed_ln(const int* __restrict__ x,
                   const float* __restrict__ t0, const float* __restrict__ t1,
                   const float* __restrict__ t2, const float* __restrict__ t3,
                   const float* __restrict__ t4, const float* __restrict__ t5,
                   const float* __restrict__ t6, const float* __restrict__ t7,
                   const float* __restrict__ t8,
                   const float* __restrict__ g, const float* __restrict__ b,
                   float* __restrict__ h, int rows) {
  int wave = (blockIdx.x * blockDim.x + threadIdx.x) >> 5;
  int lane = threadIdx.x & 31;
  if (wave >= rows) return;
  const float* tabs[9] = {t0, t1, t2, t3, t4, t5, t6, t7, t8};
  float acc[8];
#pragma unroll
  for (int j = 0; j < 8; ++j) acc[j] = 0.f;
#pragma unroll
  for (int i = 0; i < 9; ++i) {
    int idx = x[wave * 9 + i];
    const float* row = tabs[i] + (size_t)idx * HDIM;
#pragma unroll
    for (int j = 0; j < 8; ++j) acc[j] += row[lane * 8 + j];
  }
  float s = 0.f, sq = 0.f;
#pragma unroll
  for (int j = 0; j < 8; ++j) { s += acc[j]; sq += acc[j] * acc[j]; }
#pragma unroll
  for (int m = 16; m > 0; m >>= 1) { s += __shfl_xor(s, m, 32); sq += __shfl_xor(sq, m, 32); }
  float mean = s * (1.f / HDIM);
  float var  = sq * (1.f / HDIM) - mean * mean;
  float rs = rsqrtf(var + 1e-5f);
  float* orow = h + (size_t)wave * HDIM;
#pragma unroll
  for (int j = 0; j < 8; ++j) {
    int c = lane * 8 + j;
    orow[c] = (acc[j] - mean) * rs * g[c] + b[c];
  }
}

// --------------------------------------------- edge embedding sum + LN -> bf16
__global__ __launch_bounds__(256)
void edge_embed_ln(const int* __restrict__ ea,
                   const float* __restrict__ t0, const float* __restrict__ t1,
                   const float* __restrict__ t2,
                   const float* __restrict__ g, const float* __restrict__ b,
                   bf16* __restrict__ e, int rows) {
  int wave = (blockIdx.x * blockDim.x + threadIdx.x) >> 5;
  int lane = threadIdx.x & 31;
  if (wave >= rows) return;
  const float* tabs[3] = {t0, t1, t2};
  float acc[8];
#pragma unroll
  for (int j = 0; j < 8; ++j) acc[j] = 0.f;
#pragma unroll
  for (int i = 0; i < 3; ++i) {
    int idx = ea[wave * 3 + i];
    const float* row = tabs[i] + (size_t)idx * HDIM;
#pragma unroll
    for (int j = 0; j < 8; ++j) acc[j] += row[lane * 8 + j];
  }
  float s = 0.f, sq = 0.f;
#pragma unroll
  for (int j = 0; j < 8; ++j) { s += acc[j]; sq += acc[j] * acc[j]; }
#pragma unroll
  for (int m = 16; m > 0; m >>= 1) { s += __shfl_xor(s, m, 32); sq += __shfl_xor(sq, m, 32); }
  float mean = s * (1.f / HDIM);
  float var  = sq * (1.f / HDIM) - mean * mean;
  float rs = rsqrtf(var + 1e-5f);
  bf16* orow = e + (size_t)wave * HDIM;
#pragma unroll
  for (int j = 0; j < 8; ++j) {
    int c = lane * 8 + j;
    orow[c] = (bf16)((acc[j] - mean) * rs * g[c] + b[c]);
  }
}

// --------------------------- messages: agg[dst] += relu(h[src] + e)  (atomics)
__global__ __launch_bounds__(256)
void scatter_msgs(const int* __restrict__ eidx, const float* __restrict__ h,
                  const bf16* __restrict__ e, float* __restrict__ agg, int E) {
  int lane = threadIdx.x & 31;
  int wave = (blockIdx.x * blockDim.x + threadIdx.x) >> 5;
  int nwaves = (gridDim.x * blockDim.x) >> 5;
  for (int ed = wave; ed < E; ed += nwaves) {
    int s = eidx[ed];
    int d = eidx[E + ed];
    const float* hr = h + (size_t)s * HDIM;
    const bf16*  er = e + (size_t)ed * HDIM;
    float* ar = agg + (size_t)d * HDIM;
#pragma unroll
    for (int j = 0; j < 8; ++j) {
      int c = lane * 8 + j;
      float m = hr[c] + (float)er[c];
      m = fmaxf(m, 0.f);
      atomAddF(ar + c, m);
    }
  }
}

// ------------------------------- z = (1+eps[l])*h + agg, converted to bf16
__global__ void fuse_eps(const float* __restrict__ h, const float* __restrict__ agg,
                         const float* __restrict__ eps, int l,
                         bf16* __restrict__ zb, int count) {
  float s = 1.f + eps[l];
  int i = blockIdx.x * blockDim.x + threadIdx.x;
  int stride = gridDim.x * blockDim.x;
  for (; i < count; i += stride) zb[i] = (bf16)(s * h[i] + agg[i]);
}

// ----------------------------------------------------- WMMA bf16 GEMM
// C[M,Ncols] = act(A[M,K] @ B[K,Ncols] + bias), A/B bf16 row-major, f32 accum.
// Block: 256 threads = 8 waves. Block tile 128(M) x 64(N); wave tile 16 x 64.
// B tile is double-buffered in LDS, streamed with GLOBAL_LOAD_ASYNC_TO_LDS_B128
// (ASYNCcnt-tracked) so the next K-tile loads while the current one is in WMMA.
template <bool RELU, typename OutT>
__global__ __launch_bounds__(256, 2)
void gemm_bf16_wmma(const bf16* __restrict__ A, const bf16* __restrict__ B,
                    const float* __restrict__ bias, OutT* __restrict__ C,
                    int M, int K, int Ncols) {
  __shared__ __align__(16) bf16 Bs[2][32][72];  // 2 x (32K x 64N) tiles, padded rows

  const int tid  = threadIdx.x;
  const int wave = tid >> 5;
  const int lane = tid & 31;
  const int r    = lane & 15;   // A-fragment: row within wave tile
  const int hi   = lane >> 4;   // A-fragment: K half selector

  const int m0 = blockIdx.y * 128 + wave * 16;
  const int n0 = blockIdx.x * 64;

  int ar = m0 + r; if (ar > M - 1) ar = M - 1;   // clamp (ragged M)
  const size_t arow = (size_t)ar * K;

  const int brow = tid >> 3;        // 0..31 : K row of B tile
  const int bcol = (tid & 7) * 8;   // 0..56 : N col group (8 bf16 = 16B)

  // 32-bit LDS byte address of this thread's staging slot in buffer 0.
  const unsigned ldsSlot0 =
      (unsigned)(unsigned long long)(lds_bf16_t*)&Bs[0][brow][bcol];
  const unsigned ldsBufStride = (unsigned)(sizeof(bf16) * 32 * 72);

  v8f acc[4] = {};

  // Prologue: async-stage tile 0 into buffer 0.
  {
    const bf16* gB = B + (size_t)brow * Ncols + n0 + bcol;
    asm volatile("global_load_async_to_lds_b128 %0, %1, off"
                 :: "v"(ldsSlot0), "v"(gB) : "memory");
  }

  for (int k0 = 0, it = 0; k0 < K; k0 += 32, ++it) {
    const int cur = it & 1;
    // Own async stores issued for tile `it` have landed in LDS:
    asm volatile("s_wait_asynccnt 0" ::: "memory");
    // All threads' pieces of tile `it` landed; everyone finished reading the
    // other buffer in the previous iteration:
    __syncthreads();

    // Kick off tile it+1 into the spare buffer while we compute tile it.
    if (k0 + 32 < K) {
      const unsigned ldsDst = ldsSlot0 + (unsigned)(1 - cur) * ldsBufStride;
      const bf16* gB = B + (size_t)(k0 + 32 + brow) * Ncols + n0 + bcol;
      asm volatile("global_load_async_to_lds_b128 %0, %1, off"
                   :: "v"(ldsDst), "v"(gB) : "memory");
    }

    // A fragment per ISA layout: lane(r,hi) elem e<8 -> k=8*hi+e ; e>=8 -> k=16+8*hi+(e-8)
    v16bf a;
    const bf16* ap = A + arow + k0 + 8 * hi;
#pragma unroll
    for (int i = 0; i < 8; ++i) a[i] = ap[i];
#pragma unroll
    for (int i = 0; i < 8; ++i) a[8 + i] = ap[16 + i];
    if (k0 + 32 < K) __builtin_prefetch(ap + 32, 0, 3);  // next A frag -> near caches

#pragma unroll
    for (int nt = 0; nt < 4; ++nt) {
      // B fragment: lane(n=lane&15, hi) elem i<8 -> k=8*hi+i ; i>=8 -> k=16+8*hi+(i-8)
      v16bf bf;
      const bf16* bp = &Bs[cur][8 * hi][nt * 16 + r];
#pragma unroll
      for (int i = 0; i < 8; ++i) bf[i] = bp[i * 72];
#pragma unroll
      for (int i = 0; i < 8; ++i) bf[8 + i] = bp[(16 + i) * 72];
      acc[nt] = __builtin_amdgcn_wmma_f32_16x16x32_bf16(
          false, a, false, bf, (short)0, acc[nt], false, false);
    }
  }

  // Epilogue. C layout: lane(n=lane&15, mh=lane>>4), VGPR v -> M = 8*mh + v.
  const int nc = lane & 15;
  const int mh = lane >> 4;
#pragma unroll
  for (int nt = 0; nt < 4; ++nt) {
    const int col = n0 + nt * 16 + nc;
    const float bv = bias[col];
#pragma unroll
    for (int v = 0; v < 8; ++v) {
      int mm = m0 + mh * 8 + v;
      if (mm < M) {
        float xv = acc[nt][v] + bv;
        if (RELU) xv = fmaxf(xv, 0.f);
        C[(size_t)mm * Ncols + col] = (OutT)xv;
      }
    }
  }
}

// ------------------------------------------- h = relu(LayerNorm(z2; g, b))
__global__ __launch_bounds__(256)
void ln_relu_rows(const float* __restrict__ z, const float* __restrict__ g,
                  const float* __restrict__ b, float* __restrict__ out, int rows) {
  int wave = (blockIdx.x * blockDim.x + threadIdx.x) >> 5;
  int lane = threadIdx.x & 31;
  if (wave >= rows) return;
  const float* zr = z + (size_t)wave * HDIM;
  float v[8];
  float s = 0.f, sq = 0.f;
#pragma unroll
  for (int j = 0; j < 8; ++j) {
    v[j] = zr[lane * 8 + j];
    s += v[j]; sq += v[j] * v[j];
  }
#pragma unroll
  for (int m = 16; m > 0; m >>= 1) { s += __shfl_xor(s, m, 32); sq += __shfl_xor(sq, m, 32); }
  float mean = s * (1.f / HDIM);
  float var  = sq * (1.f / HDIM) - mean * mean;
  float rs = rsqrtf(var + 1e-5f);
  float* orow = out + (size_t)wave * HDIM;
#pragma unroll
  for (int j = 0; j < 8; ++j) {
    int c = lane * 8 + j;
    float y = (v[j] - mean) * rs * g[c] + b[c];
    orow[c] = fmaxf(y, 0.f);
  }
}

// ----------------------------------------------------- global mean pool
__global__ __launch_bounds__(256)
void pool_scatter(const int* __restrict__ bid, const float* __restrict__ h,
                  float* __restrict__ gsum, float* __restrict__ gcnt, int rows) {
  int wave = (blockIdx.x * blockDim.x + threadIdx.x) >> 5;
  int lane = threadIdx.x & 31;
  if (wave >= rows) return;
  int g = bid[wave];
  const float* hr = h + (size_t)wave * HDIM;
  float* gr = gsum + (size_t)g * HDIM;
#pragma unroll
  for (int j = 0; j < 8; ++j) {
    int c = lane * 8 + j;
    atomAddF(gr + c, hr[c]);
  }
  if (lane == 0) atomAddF(gcnt + g, 1.f);
}

__global__ __launch_bounds__(256)
void pool_div(const float* __restrict__ gsum, const float* __restrict__ gcnt,
              bf16* __restrict__ gb, int rows) {
  int wave = (blockIdx.x * blockDim.x + threadIdx.x) >> 5;
  int lane = threadIdx.x & 31;
  if (wave >= rows) return;
  float inv = 1.f / fmaxf(gcnt[wave], 1.f);
  const float* sr = gsum + (size_t)wave * HDIM;
  bf16* orow = gb + (size_t)wave * HDIM;
#pragma unroll
  for (int j = 0; j < 8; ++j) {
    int c = lane * 8 + j;
    orow[c] = (bf16)(sr[c] * inv);
  }
}

// ------------------------------------- out = out / max(||out||, 1e-12), row-wise
__global__ __launch_bounds__(256)
void l2_normalize(float* __restrict__ out, int rows) {
  int wave = (blockIdx.x * blockDim.x + threadIdx.x) >> 5;
  int lane = threadIdx.x & 31;
  if (wave >= rows) return;
  float* orow = out + (size_t)wave * HDIM;
  float v[8];
  float sq = 0.f;
#pragma unroll
  for (int j = 0; j < 8; ++j) { v[j] = orow[lane * 8 + j]; sq += v[j] * v[j]; }
#pragma unroll
  for (int m = 16; m > 0; m >>= 1) sq += __shfl_xor(sq, m, 32);
  float inv = 1.f / fmaxf(sqrtf(sq), 1e-12f);
#pragma unroll
  for (int j = 0; j < 8; ++j) orow[lane * 8 + j] = v[j] * inv;
}

// ============================================================== host driver
extern "C" void kernel_launch(void* const* d_in, const int* in_sizes, int n_in,
                              void* d_out, int out_size, void* d_ws, size_t ws_size,
                              hipStream_t stream) {
  (void)in_sizes; (void)n_in; (void)out_size; (void)ws_size;
  const int N = 150000, E = 300000, G = 5000, H = 256, L = 4;

  const int*   x     = (const int*)d_in[0];
  const int*   eidx  = (const int*)d_in[1];
  const int*   eattr = (const int*)d_in[2];
  const int*   bid   = (const int*)d_in[3];
  const float* nt[9];
  for (int i = 0; i < 9; ++i) nt[i] = (const float*)d_in[4 + i];
  const float* et[3];
  for (int i = 0; i < 3; ++i) et[i] = (const float*)d_in[13 + i];
  const float* nln_g = (const float*)d_in[16];
  const float* nln_b = (const float*)d_in[17];
  const float* eln_g = (const float*)d_in[18];
  const float* eln_b = (const float*)d_in[19];
  const float* W1  = (const float*)d_in[20];
  const float* b1  = (const float*)d_in[21];
  const float* W2  = (const float*)d_in[22];
  const float* b2  = (const float*)d_in[23];
  const float* eps = (const float*)d_in[24];
  const float* lng = (const float*)d_in[25];
  const float* lnb = (const float*)d_in[26];
  const float* pW1 = (const float*)d_in[27];
  const float* pb1 = (const float*)d_in[28];
  const float* pW2 = (const float*)d_in[29];
  const float* pb2 = (const float*)d_in[30];
  float* out = (float*)d_out;

  // -------- workspace carve-out (256B aligned slices)
  char* ws = (char*)d_ws;
  size_t off = 0;
  auto carve = [&](size_t bytes) -> void* {
    void* p = ws + off;
    off += (bytes + 255) & ~(size_t)255;
    return p;
  };
  float* h    = (float*)carve((size_t)N * H * 4);
  bf16*  e    = (bf16*) carve((size_t)E * H * 2);
  float* agg  = (float*)carve((size_t)N * H * 4);
  bf16*  zb   = (bf16*) carve((size_t)N * H * 2);
  bf16*  mid  = (bf16*) carve((size_t)N * 2 * H * 2);
  float* z2   = (float*)carve((size_t)N * H * 4);
  bf16*  w1b  = (bf16*) carve((size_t)L * H * 2 * H * 2);
  bf16*  w2b  = (bf16*) carve((size_t)L * 2 * H * H * 2);
  bf16*  pw1b = (bf16*) carve((size_t)H * H * 2);
  bf16*  pw2b = (bf16*) carve((size_t)H * H * 2);
  float* gsum = (float*)carve((size_t)G * H * 4);
  float* gcnt = (float*)carve((size_t)G * 4);
  bf16*  gb   = (bf16*) carve((size_t)G * H * 2);
  bf16*  gmid = (bf16*) carve((size_t)G * H * 2);

  // -------- convert weights f32 -> bf16
  cvt_f32_bf16<<<2048, 256, 0, stream>>>(W1,  w1b,  L * H * 2 * H);
  cvt_f32_bf16<<<2048, 256, 0, stream>>>(W2,  w2b,  L * 2 * H * H);
  cvt_f32_bf16<<<256,  256, 0, stream>>>(pW1, pw1b, H * H);
  cvt_f32_bf16<<<256,  256, 0, stream>>>(pW2, pw2b, H * H);

  // -------- embeddings + LN
  node_embed_ln<<<N / 8, 256, 0, stream>>>(x, nt[0], nt[1], nt[2], nt[3], nt[4],
                                           nt[5], nt[6], nt[7], nt[8],
                                           nln_g, nln_b, h, N);
  edge_embed_ln<<<E / 8, 256, 0, stream>>>(eattr, et[0], et[1], et[2],
                                           eln_g, eln_b, e, E);

  const int gy_nodes = (N + 127) / 128;
  for (int l = 0; l < L; ++l) {
    zero_f32<<<4096, 256, 0, stream>>>(agg, N * H / 4);
    scatter_msgs<<<2048, 256, 0, stream>>>(eidx, h, e, agg, E);
    fuse_eps<<<4096, 256, 0, stream>>>(h, agg, eps, l, zb, N * H);
    // z1 = relu(z @ W1[l] + b1[l])   [N,256] x [256,512]
    gemm_bf16_wmma<true, bf16><<<dim3(2 * H / 64, gy_nodes), 256, 0, stream>>>(
        zb, w1b + (size_t)l * H * 2 * H, b1 + (size_t)l * 2 * H, mid, N, H, 2 * H);
    // z2 = z1 @ W2[l] + b2[l]        [N,512] x [512,256]
    gemm_bf16_wmma<false, float><<<dim3(H / 64, gy_nodes), 256, 0, stream>>>(
        mid, w2b + (size_t)l * 2 * H * H, b2 + (size_t)l * H, z2, N, 2 * H, H);
    // h = relu(LN(z2))
    ln_relu_rows<<<N / 8, 256, 0, stream>>>(z2, lng + (size_t)l * H, lnb + (size_t)l * H, h, N);
  }

  // -------- global mean pool + head MLP + normalize
  zero_f32<<<2048, 256, 0, stream>>>(gsum, G * H / 4);
  zero_f32<<<16,   256, 0, stream>>>(gcnt, G / 4);
  pool_scatter<<<N / 8, 256, 0, stream>>>(bid, h, gsum, gcnt, N);
  pool_div<<<(G + 7) / 8, 256, 0, stream>>>(gsum, gcnt, gb, G);

  const int gy_graphs = (G + 127) / 128;
  gemm_bf16_wmma<true, bf16><<<dim3(H / 64, gy_graphs), 256, 0, stream>>>(
      gb, pw1b, pb1, gmid, G, H, H);
  gemm_bf16_wmma<false, float><<<dim3(H / 64, gy_graphs), 256, 0, stream>>>(
      gmid, pw2b, pb2, out, G, H, H);
  l2_normalize<<<(G + 7) / 8, 256, 0, stream>>>(out, G);
}
